// Informer_19344532701757
// MI455X (gfx1250) — matrix operand order repaired and checked by reading, not compiled, not run on hardware
//
#include <hip/hip_runtime.h>
#include <math.h>

// ---------------- types for WMMA ----------------
typedef __attribute__((ext_vector_type(16))) __bf16       v16bf;
typedef __attribute__((ext_vector_type(8)))  float        v8f;
typedef __attribute__((ext_vector_type(8)))  unsigned int u32x8;

__device__ __forceinline__ unsigned short f2bf(float f) {
    unsigned u = __builtin_bit_cast(unsigned, f);
    u += 0x7FFFu + ((u >> 16) & 1u);           // round-to-nearest-even
    return (unsigned short)(u >> 16);
}

// ---------------- Threefry-2x32 (host+device) ----------------
#define TF_RR(r) do { x0 += x1; x1 = (x1 << (r)) | (x1 >> (32 - (r))); x1 ^= x0; } while (0)
__host__ __device__ inline void threefry2x32(unsigned k0, unsigned k1,
                                             unsigned c0, unsigned c1,
                                             unsigned* r0, unsigned* r1) {
    unsigned ks2 = 0x1BD11BDAu ^ k0 ^ k1;
    unsigned x0 = c0 + k0, x1 = c1 + k1;
    TF_RR(13); TF_RR(15); TF_RR(26); TF_RR(6);  x0 += k1;  x1 += ks2 + 1u;
    TF_RR(17); TF_RR(29); TF_RR(16); TF_RR(24); x0 += ks2; x1 += k0 + 2u;
    TF_RR(13); TF_RR(15); TF_RR(26); TF_RR(6);  x0 += k0;  x1 += k1 + 3u;
    TF_RR(17); TF_RR(29); TF_RR(16); TF_RR(24); x0 += k1;  x1 += ks2 + 4u;
    TF_RR(13); TF_RR(15); TF_RR(26); TF_RR(6);  x0 += ks2; x1 += k0 + 5u;
    *r0 = x0; *r1 = x1;
}

// ============================================================
//  Generic batched WMMA GEMM:  C[m,n] = sum_k A[m,k]*B[n,k] + bias[n]
//  A element  = A[offA + m*lda_m + k*lda_k]
//  B element  = B[offB + n*ldb_n + k*ldb_k]
//  batch z = blockIdx.z -> (b = z/batH, h = z%batH), offsets via strides.
//  act: 0 = none, 1 = exact GELU
//  Fast path: contiguous-K, aligned row stride, interior tile ->
//             float4 global loads + next-tile prefetch.
// ============================================================
#define BMT 64
#define BNT 128
#define BKT 32
#define LDP (BKT + 8)   // row stride (ushorts) -> 80B, 16B aligned

__global__ __launch_bounds__(256) void k_gemm(
    const float* __restrict__ A, long sAb, long sAh, int lda_m, int lda_k,
    const float* __restrict__ B, long sBb, long sBh, int ldb_n, int ldb_k,
    const float* __restrict__ bias,
    float* __restrict__ C, long sCb, long sCh, int ldc,
    int M, int N, int K, int act, int batH)
{
    __shared__ __align__(16) unsigned short As[BMT][LDP];
    __shared__ __align__(16) unsigned short Bs[BNT][LDP];

    const int z  = blockIdx.z;
    const int bb = z / batH, hh = z % batH;
    const float* Ab = A + bb * sAb + hh * sAh;
    const float* Bb = B + bb * sBb + hh * sBh;
    float*       Cb = C + bb * sCb + hh * sCh;

    const int bM = blockIdx.y * BMT, bN = blockIdx.x * BNT;
    const int t = threadIdx.x;
    const int wave = t >> 5, lane = t & 31;
    const int wm = wave >> 2, wn = wave & 3;     // 2 x 4 wave grid
    const int lr = lane & 15, lh = lane >> 4;

    v8f acc[2][2] = {};

    const int ar = t >> 2, ac = (t & 3) * 8;     // A: 64 rows x 4 threads x 8 elems
    const int br = t >> 1, bc = (t & 1) * 16;    // B: 128 rows x 2 threads x 16 elems

    const bool fastA = (lda_k == 1) && ((lda_m & 3) == 0) && (bM + BMT <= M);
    const bool fastB = (ldb_k == 1) && ((ldb_n & 3) == 0) && (bN + BNT <= N);

    for (int kt = 0; kt < K; kt += BKT) {
        const bool kfull = (kt + BKT <= K);
        {   // ---- stage A tile (fp32 -> bf16) ----
            long m = (long)bM + ar;
            if (fastA && kfull) {
                const float* src = Ab + m * (long)lda_m + kt + ac;
                float4 f0 = ((const float4*)src)[0];
                float4 f1 = ((const float4*)src)[1];
                As[ar][ac + 0] = f2bf(f0.x); As[ar][ac + 1] = f2bf(f0.y);
                As[ar][ac + 2] = f2bf(f0.z); As[ar][ac + 3] = f2bf(f0.w);
                As[ar][ac + 4] = f2bf(f1.x); As[ar][ac + 5] = f2bf(f1.y);
                As[ar][ac + 6] = f2bf(f1.z); As[ar][ac + 7] = f2bf(f1.w);
                if (kt + 2 * BKT <= K) __builtin_prefetch(src + BKT, 0, 0);
            } else {
                #pragma unroll
                for (int j = 0; j < 8; ++j) {
                    int k = ac + j; float val = 0.f;
                    if (m < M && (kt + k) < K)
                        val = Ab[m * (long)lda_m + (long)(kt + k) * lda_k];
                    As[ar][k] = f2bf(val);
                }
            }
            // ---- stage B tile ----
            long n = (long)bN + br;
            if (fastB && kfull) {
                const float* src = Bb + n * (long)ldb_n + kt + bc;
                float4 f0 = ((const float4*)src)[0];
                float4 f1 = ((const float4*)src)[1];
                float4 f2 = ((const float4*)src)[2];
                float4 f3 = ((const float4*)src)[3];
                Bs[br][bc +  0] = f2bf(f0.x); Bs[br][bc +  1] = f2bf(f0.y);
                Bs[br][bc +  2] = f2bf(f0.z); Bs[br][bc +  3] = f2bf(f0.w);
                Bs[br][bc +  4] = f2bf(f1.x); Bs[br][bc +  5] = f2bf(f1.y);
                Bs[br][bc +  6] = f2bf(f1.z); Bs[br][bc +  7] = f2bf(f1.w);
                Bs[br][bc +  8] = f2bf(f2.x); Bs[br][bc +  9] = f2bf(f2.y);
                Bs[br][bc + 10] = f2bf(f2.z); Bs[br][bc + 11] = f2bf(f2.w);
                Bs[br][bc + 12] = f2bf(f3.x); Bs[br][bc + 13] = f2bf(f3.y);
                Bs[br][bc + 14] = f2bf(f3.z); Bs[br][bc + 15] = f2bf(f3.w);
                if (kt + 2 * BKT <= K) __builtin_prefetch(src + BKT, 0, 0);
            } else {
                #pragma unroll
                for (int j = 0; j < 16; ++j) {
                    int k = bc + j; float val = 0.f;
                    if (n < N && (kt + k) < K)
                        val = Bb[n * (long)ldb_n + (long)(kt + k) * ldb_k];
                    Bs[br][k] = f2bf(val);
                }
            }
        }
        __syncthreads();

        v16bf afrag[2], bfrag[2];
        #pragma unroll
        for (int i = 0; i < 2; ++i) {                  // A 16x32 frag (ISA 7.12.2)
            const unsigned* p = (const unsigned*)&As[wm * 32 + i * 16 + lr][0];
            u32x8 ua;
            #pragma unroll
            for (int q = 0; q < 4; ++q) ua[q]     = p[lh * 4 + q];      // K = lh*8 .. +7
            #pragma unroll
            for (int q = 0; q < 4; ++q) ua[4 + q] = p[8 + lh * 4 + q];  // K = 16+lh*8 .. +7
            afrag[i] = __builtin_bit_cast(v16bf, ua);
        }
        #pragma unroll
        for (int j = 0; j < 2; ++j) {                  // B 32x16 frag
            const unsigned* p = (const unsigned*)&Bs[wn * 32 + j * 16 + lr][0];
            u32x8 ub;
            #pragma unroll
            for (int q = 0; q < 4; ++q) ub[q]     = p[lh * 4 + q];      // K = lh*8 .. +7
            #pragma unroll
            for (int q = 0; q < 4; ++q) ub[4 + q] = p[8 + lh * 4 + q];  // K = 16+lh*8 .. +7
            bfrag[j] = __builtin_bit_cast(v16bf, ub);
        }
        #pragma unroll
        for (int i = 0; i < 2; ++i)
            #pragma unroll
            for (int j = 0; j < 2; ++j)
                acc[i][j] = __builtin_amdgcn_wmma_f32_16x16x32_bf16(
                    false, afrag[i], false, bfrag[j], (short)0, acc[i][j], false, false);
        __syncthreads();
    }

    #pragma unroll
    for (int i = 0; i < 2; ++i)
        #pragma unroll
        for (int j = 0; j < 2; ++j) {
            int n = bN + wn * 32 + j * 16 + lr;
            if (n >= N) continue;
            float bv = bias ? bias[n] : 0.f;
            #pragma unroll
            for (int v = 0; v < 8; ++v) {
                int m = bM + wm * 32 + i * 16 + v + 8 * lh;
                if (m >= M) continue;
                float val = acc[i][j][v] + bv;
                if (act == 1) val = 0.5f * val * (1.f + erff(val * 0.70710678f));
                Cb[(long)m * ldc + n] = val;
            }
        }
}

// ---------------- positional embedding ----------------
__global__ void k_posembed(float* x, int Bn, int L) {
    long i = (long)blockIdx.x * 256 + threadIdx.x;
    long tot = (long)Bn * L * 512; if (i >= tot) return;
    int d = (int)(i & 511); long bl = i >> 9; int l = (int)(bl % L);
    int j = d >> 1;
    float ang = (float)l * expf(-(float)(2 * j) * (logf(10000.f) / 512.f));
    x[i] = (d & 1) ? cosf(ang) : sinf(ang);
}

// ---------------- layernorm (optional fused residual) ----------------
__global__ __launch_bounds__(256) void k_layernorm(const float* __restrict__ x,
                                                   const float* __restrict__ res,
                                                   const float* __restrict__ g,
                                                   const float* __restrict__ b,
                                                   float* __restrict__ y) {
    long row = blockIdx.x; int t = threadIdx.x;
    __shared__ float s1[256], s2[256];
    const float* xr = x + row * 512;
    float v0 = xr[t]       + (res ? res[row * 512 + t]       : 0.f);
    float v1 = xr[t + 256] + (res ? res[row * 512 + t + 256] : 0.f);
    s1[t] = v0 + v1; s2[t] = v0 * v0 + v1 * v1;
    __syncthreads();
    for (int s = 128; s > 0; s >>= 1) {
        if (t < s) { s1[t] += s1[t + s]; s2[t] += s2[t + s]; }
        __syncthreads();
    }
    float m = s1[0] * (1.f / 512.f);
    float var = s2[0] * (1.f / 512.f) - m * m;
    float inv = rsqrtf(var + 1e-5f);
    y[row * 512 + t]       = (v0 - m) * inv * g[t]       + b[t];
    y[row * 512 + t + 256] = (v1 - m) * inv * g[t + 256] + b[t + 256];
}

// ---------------- randint sampling (JAX-style threefry bits) ----------------
__global__ void k_randint(int* idx, int n, int span, unsigned k0, unsigned k1) {
    int i = blockIdx.x * 256 + threadIdx.x; if (i >= n) return;
    unsigned o0, o1, hi, lo;
    threefry2x32(k0, k1, (unsigned)i, (unsigned)(i + n), &o0, &o1);
    hi = o0; lo = o1;
    unsigned sp = (unsigned)span;
    unsigned mult = ((65536u % sp) * (65536u % sp)) % sp;
    unsigned off = hi * mult + lo;
    idx[i] = (int)(off % sp);
}

// ---------------- sparsity measure M ----------------
__global__ void k_qks_M(const float* __restrict__ q, const float* __restrict__ k,
                        const int* __restrict__ idx, float* __restrict__ Mv,
                        int Bn, int L, int U) {
    long i = (long)blockIdx.x * 256 + threadIdx.x;
    long tot = (long)Bn * 8 * L; if (i >= tot) return;
    int l = (int)(i % L); long zh = i / L; int h = (int)(zh % 8); int b = (int)(zh / 8);
    const float* qr = q + ((long)b * L + l) * 512 + h * 64;
    float mx = -INFINITY, sm = 0.f;
    for (int uu = 0; uu < U; ++uu) {
        int ki = idx[l * U + uu];
        const float* kr = k + ((long)b * L + ki) * 512 + h * 64;
        float d = 0.f;
        for (int e = 0; e < 64; ++e) d += qr[e] * kr[e];
        mx = fmaxf(mx, d); sm += d;
    }
    Mv[i] = mx - sm / (float)L;
}

// ---------------- per-(b,h) iterative top-u ----------------
__global__ __launch_bounds__(256) void k_topk(const float* __restrict__ M, int* __restrict__ mtop,
                                              int L, int u) {
    int z = blockIdx.x, t = threadIdx.x;
    __shared__ float sv[4096];
    __shared__ float rv[256]; __shared__ int ri[256];
    for (int i = t; i < L; i += 256) sv[i] = M[(long)z * L + i];
    __syncthreads();
    for (int it = 0; it < u; ++it) {
        float bv = -INFINITY; int bi = 0x7FFFFFFF;
        for (int i = t; i < L; i += 256) {
            float v = sv[i];
            if (v > bv || (v == bv && i < bi)) { bv = v; bi = i; }
        }
        rv[t] = bv; ri[t] = bi; __syncthreads();
        for (int s = 128; s > 0; s >>= 1) {
            if (t < s) {
                if (rv[t + s] > rv[t] || (rv[t + s] == rv[t] && ri[t + s] < ri[t])) {
                    rv[t] = rv[t + s]; ri[t] = ri[t + s];
                }
            }
            __syncthreads();
        }
        if (t == 0) { mtop[z * u + it] = ri[0]; sv[ri[0]] = -INFINITY; }
        __syncthreads();
    }
}

// ---------------- scores for selected queries (scale 1/sqrt(64), causal opt.) --
__global__ void k_scores(const float* __restrict__ q, const float* __restrict__ k,
                         const int* __restrict__ mtop, float* __restrict__ S,
                         int Bn, int L, int u, int masked) {
    long i = (long)blockIdx.x * 256 + threadIdx.x;
    long tot = (long)Bn * 8 * u * L; if (i >= tot) return;
    int s = (int)(i % L); long r = i / L; int uu = (int)(r % u); long z = r / u;
    int h = (int)(z % 8), b = (int)(z / 8);
    int m = mtop[z * u + uu];
    float val;
    if (masked && s > m) val = -INFINITY;
    else {
        const float* qr = q + ((long)b * L + m) * 512 + h * 64;
        const float* kr = k + ((long)b * L + s) * 512 + h * 64;
        float d = 0.f;
        for (int e = 0; e < 64; ++e) d += qr[e] * kr[e];
        val = d * 0.125f;
    }
    S[i] = val;
}

// ---------------- row softmax ----------------
__global__ __launch_bounds__(256) void k_softmax(float* __restrict__ S, int cols, float scale) {
    long row = blockIdx.x; float* r = S + row * (long)cols;
    int t = threadIdx.x;
    __shared__ float red[256];
    float mx = -INFINITY;
    for (int i = t; i < cols; i += 256) mx = fmaxf(mx, r[i] * scale);
    red[t] = mx; __syncthreads();
    for (int s = 128; s > 0; s >>= 1) { if (t < s) red[t] = fmaxf(red[t], red[t + s]); __syncthreads(); }
    mx = red[0]; __syncthreads();
    float sm = 0.f;
    for (int i = t; i < cols; i += 256) sm += expf(r[i] * scale - mx);
    red[t] = sm; __syncthreads();
    for (int s = 128; s > 0; s >>= 1) { if (t < s) red[t] += red[t + s]; __syncthreads(); }
    float inv = 1.f / red[0]; __syncthreads();
    for (int i = t; i < cols; i += 256) r[i] = expf(r[i] * scale - mx) * inv;
}

// ---------------- context init: mean-of-V (broadcast) ----------------
__global__ __launch_bounds__(256) void k_vmean(const float* __restrict__ v, float* __restrict__ mean,
                                               int Bn, int L) {
    int z = blockIdx.x;                         // (b*8+h)*64+d
    int d = z & 63, h = (z >> 6) & 7, b = z >> 9;
    int t = threadIdx.x;
    __shared__ float red[256];
    float s = 0.f;
    for (int l = t; l < L; l += 256) s += v[((long)b * L + l) * 512 + h * 64 + d];
    red[t] = s; __syncthreads();
    for (int sh = 128; sh > 0; sh >>= 1) { if (t < sh) red[t] += red[t + sh]; __syncthreads(); }
    if (t == 0) mean[z] = red[0] / (float)L;
}
__global__ void k_ctx_bcast(float* ctx, const float* mean, int Bn, int L) {
    long i = (long)blockIdx.x * 256 + threadIdx.x;
    long tot = (long)Bn * L * 512; if (i >= tot) return;
    long bl = i >> 9; int b = (int)(bl / L);
    ctx[i] = mean[b * 512 + (int)(i & 511)];
}
// ---------------- context init: cumsum-of-V (masked/decoder) ----------------
__global__ void k_cumsum(const float* __restrict__ v, float* __restrict__ ctx, int Bn, int L) {
    int i = blockIdx.x * 256 + threadIdx.x;
    if (i >= Bn * 512) return;
    int hd = i & 511, b = i >> 9;
    float run = 0.f;
    for (int l = 0; l < L; ++l) {
        long o = ((long)b * L + l) * 512 + hd;
        run += v[o]; ctx[o] = run;
    }
}

// ---------------- update rows and scatter ----------------
__global__ void k_upd(const float* __restrict__ P, const float* __restrict__ v,
                      float* __restrict__ upd, int Bn, int L, int u) {
    long i = (long)blockIdx.x * 256 + threadIdx.x;
    long tot = (long)Bn * 8 * u * 64; if (i >= tot) return;
    int d = (int)(i & 63); long r = i >> 6; long z = r / u;
    int h = (int)(z % 8), b = (int)(z / 8);
    const float* pr = P + r * (long)L;
    float acc = 0.f;
    for (int s = 0; s < L; ++s) acc += pr[s] * v[((long)b * L + s) * 512 + h * 64 + d];
    upd[i] = acc;
}
__global__ void k_scatter(const float* __restrict__ upd, const int* __restrict__ mtop,
                          float* __restrict__ ctx, int Bn, int L, int u) {
    long i = (long)blockIdx.x * 256 + threadIdx.x;
    long tot = (long)Bn * 8 * u * 64; if (i >= tot) return;
    int d = (int)(i & 63); long r = i >> 6; int uu = (int)(r % u); long z = r / u;
    int h = (int)(z % 8), b = (int)(z / 8);
    int l = mtop[z * u + uu];
    ctx[((long)b * L + l) * 512 + h * 64 + d] = upd[i];
}

// ---------------- decoder "mix" transpose ----------------
__global__ void k_mix(const float* __restrict__ ctx, float* __restrict__ out, int Bn, int L) {
    long i = (long)blockIdx.x * 256 + threadIdx.x;
    long tot = (long)Bn * L * 512; if (i >= tot) return;
    long n = i % ((long)L * 512); int b = (int)(i / ((long)L * 512));
    int h = (int)(n / ((long)L * 64)); long rem = n % ((long)L * 64);
    int l = (int)(rem / 64), d = (int)(rem % 64);
    out[i] = ctx[((long)b * L + l) * 512 + h * 64 + d];
}

// ---------------- conv path ----------------
__global__ void k_im2col(const float* __restrict__ x, float* __restrict__ col, int Bn, int L) {
    long i = (long)blockIdx.x * 256 + threadIdx.x;
    long tot = (long)Bn * (L + 2) * 1536; if (i >= tot) return;
    int kf = (int)(i % 1536); long bt = i / 1536;
    int t = (int)(bt % (L + 2)); int b = (int)(bt / (L + 2));
    int ci = kf / 3, kk = kf % 3;
    int src = t + kk - 2; src = ((src % L) + L) % L;
    col[i] = x[((long)b * L + src) * 512 + ci];
}
__global__ __launch_bounds__(256) void k_bn_stats(const float* __restrict__ y,
                                                  float* __restrict__ mean, float* __restrict__ var,
                                                  int ntok) {
    int c = blockIdx.x, t = threadIdx.x;
    __shared__ float r1[256], r2[256];
    float s = 0.f, s2 = 0.f;
    for (int i = t; i < ntok; i += 256) { float v = y[(long)i * 512 + c]; s += v; s2 += v * v; }
    r1[t] = s; r2[t] = s2; __syncthreads();
    for (int sh = 128; sh > 0; sh >>= 1) {
        if (t < sh) { r1[t] += r1[t + sh]; r2[t] += r2[t + sh]; }
        __syncthreads();
    }
    if (t == 0) { float m = r1[0] / ntok; mean[c] = m; var[c] = r2[0] / ntok - m * m; }
}
__global__ void k_bn_elu(float* __restrict__ y, const float* mean, const float* var,
                         const float* g, const float* be, int ntok) {
    long i = (long)blockIdx.x * 256 + threadIdx.x;
    long tot = (long)ntok * 512; if (i >= tot) return;
    int c = (int)(i & 511);
    float v = (y[i] - mean[c]) * rsqrtf(var[c] + 1e-5f) * g[c] + be[c];
    y[i] = v > 0.f ? v : expm1f(v);
}
__global__ void k_maxpool(const float* __restrict__ y, float* __restrict__ out,
                          int Bn, int Lc, int Lout) {
    long i = (long)blockIdx.x * 256 + threadIdx.x;
    long tot = (long)Bn * Lout * 512; if (i >= tot) return;
    int c = (int)(i & 511); long bi = i >> 9;
    int o = (int)(bi % Lout); int b = (int)(bi / Lout);
    float m = -INFINITY;
    for (int w = 0; w < 3; ++w) {
        int j = 2 * o + w - 1;
        if (j >= 0 && j < Lc) m = fmaxf(m, y[((long)b * Lc + j) * 512 + c]);
    }
    out[i] = m;
}

// ---------------- final tiny projection (N=7) ----------------
__global__ void k_proj(const float* __restrict__ x, const float* __restrict__ pw,
                       const float* __restrict__ pb, float* __restrict__ out, int Bn) {
    int i = blockIdx.x * 256 + threadIdx.x;
    int tot = Bn * 512 * 7; if (i >= tot) return;
    int c = i % 7; int r = i / 7; int tok = r % 512; int b = r / 512;
    const float* xr = x + ((long)b * 1024 + 512 + tok) * 512;
    float acc = pb[c];
    for (int k = 0; k < 512; ++k) acc += xr[k] * pw[c * 512 + k];
    out[i] = acc;
}

// ============================================================
//  Host orchestration
// ============================================================
static inline unsigned gblk(long n) { return (unsigned)((n + 255) / 256); }

static void gemm(hipStream_t st,
                 const float* A, long sAb, long sAh, int lda_m, int lda_k,
                 const float* B, long sBb, long sBh, int ldb_n, int ldb_k,
                 const float* bias, float* C, long sCb, long sCh, int ldc,
                 int M, int N, int K, int act, int batB, int batH) {
    dim3 g((N + BNT - 1) / BNT, (M + BMT - 1) / BMT, batB * batH);
    k_gemm<<<g, 256, 0, st>>>(A, sAb, sAh, lda_m, lda_k, B, sBb, sBh, ldb_n, ldb_k,
                              bias, C, sCb, sCh, ldc, M, N, K, act, batH);
}

// workspace layout (float offsets), Bn = 1  (WO_* to avoid POSIX off_t clash)
#define ABUF   2100000L
#define BIGSZ  8500000L
static const long WO_X = 0, WO_Q = ABUF, WO_K = 2 * ABUF, WO_V = 3 * ABUF;
static const long WO_T = 4 * ABUF, WO_CTX = 5 * ABUF, WO_ENC = 6 * ABUF;
static const long WO_BIG  = 7 * ABUF;
static const long WO_M    = WO_BIG + BIGSZ;
static const long WO_IDX  = WO_M + 40000;
static const long WO_MTOP = WO_IDX + 190000;
static const long WO_UPD  = WO_MTOP + 1000;
static const long WO_MEAN = WO_UPD + 25000;

static void prob_attn_run(hipStream_t st, float* W,
                          const float* q, const float* k, const float* v, float* ctx,
                          int Bn, int L, int masked, unsigned kk0, unsigned kk1) {
    int U = 5 * (int)ceil(log((double)L)); if (U > L) U = L;
    int u = U;
    int n = L * U;
    int* idx  = (int*)(W + WO_IDX);
    int* mtop = (int*)(W + WO_MTOP);
    k_randint<<<gblk(n), 256, 0, st>>>(idx, n, L, kk0, kk1);
    long BH = (long)Bn * 8;
    k_qks_M<<<gblk(BH * L), 256, 0, st>>>(q, k, idx, W + WO_M, Bn, L, U);
    k_topk<<<(unsigned)BH, 256, 0, st>>>(W + WO_M, mtop, L, u);
    long rows = BH * u;
    k_scores<<<gblk(rows * L), 256, 0, st>>>(q, k, mtop, W + WO_BIG, Bn, L, u, masked);
    k_softmax<<<(unsigned)rows, 256, 0, st>>>(W + WO_BIG, L, 1.f);
    if (masked) {
        k_cumsum<<<gblk(Bn * 512), 256, 0, st>>>(v, ctx, Bn, L);
    } else {
        k_vmean<<<(unsigned)(BH * 64), 256, 0, st>>>(v, W + WO_MEAN, Bn, L);
        k_ctx_bcast<<<gblk((long)Bn * L * 512), 256, 0, st>>>(ctx, W + WO_MEAN, Bn, L);
    }
    k_upd<<<gblk(rows * 64), 256, 0, st>>>(W + WO_BIG, v, W + WO_UPD, Bn, L, u);
    k_scatter<<<gblk(rows * 64), 256, 0, st>>>(W + WO_UPD, mtop, ctx, Bn, L, u);
}

extern "C" void kernel_launch(void* const* d_in, const int* in_sizes, int n_in,
                              void* d_out, int out_size, void* d_ws, size_t ws_size,
                              hipStream_t stream) {
    (void)in_sizes; (void)n_in; (void)out_size; (void)ws_size;
    hipStream_t st = stream;
    float* W = (float*)d_ws;
    auto in = [&](int i) { return (const float*)d_in[i]; };

    // param index map (setup_inputs dict order):
    // 0 x_enc, 1 x_dec, enc layer i @ 2+16i: wq bq wk bk wv bv wo bo w1 b1 w2 b2 n1g n1b n2g n2b
    // conv i @ 50+4i: w b g be
    // dec layer i @ 58+26i: sattn(8) cattn(8) w1 b1 w2 b2 n1g n1b n2g n2b n3g n3b
    // tail @110: eng enb dng dnb pw pb
    const int Bn = 1;   // pos_embed(...)[None] -> effective batch 1 (x_enc values unused)
    float* X   = W + WO_X;
    float* Q   = W + WO_Q;
    float* Kb  = W + WO_K;
    float* Vb  = W + WO_V;
    float* T   = W + WO_T;
    float* CTX = W + WO_CTX;
    float* ENC = W + WO_ENC;
    float* BG  = W + WO_BIG;

    // ---------------- encoder ----------------
    int L = 4096;
    k_posembed<<<gblk((long)Bn * L * 512), 256, 0, st>>>(X, Bn, L);
    for (int li = 0; li < 3; ++li) {
        const int p = 2 + 16 * li;
        int ntok = Bn * L;
        gemm(st, X, 0, 0, 512, 1, in(p + 0), 0, 0, 512, 1, in(p + 1), Q,  0, 0, 512, ntok, 512, 512, 0, 1, 1);
        gemm(st, X, 0, 0, 512, 1, in(p + 2), 0, 0, 512, 1, in(p + 3), Kb, 0, 0, 512, ntok, 512, 512, 0, 1, 1);
        gemm(st, X, 0, 0, 512, 1, in(p + 4), 0, 0, 512, 1, in(p + 5), Vb, 0, 0, 512, ntok, 512, 512, 0, 1, 1);
        unsigned kk0, kk1; threefry2x32(0u, 42u, 0u, (unsigned)li, &kk0, &kk1);  // fold_in(base, li)
        prob_attn_run(st, W, Q, Kb, Vb, CTX, Bn, L, 0, kk0, kk1);
        gemm(st, CTX, 0, 0, 512, 1, in(p + 6), 0, 0, 512, 1, in(p + 7), T, 0, 0, 512, ntok, 512, 512, 0, 1, 1);
        k_layernorm<<<(unsigned)ntok, 256, 0, st>>>(X, T, in(p + 12), in(p + 13), X);
        gemm(st, X,  0, 0, 512,  1, in(p + 8),  0, 0, 512,  1, in(p + 9),  BG, 0, 0, 2048, ntok, 2048, 512, 1, 1, 1);
        gemm(st, BG, 0, 0, 2048, 1, in(p + 10), 0, 0, 2048, 1, in(p + 11), T,  0, 0, 512,  ntok, 512, 2048, 0, 1, 1);
        k_layernorm<<<(unsigned)ntok, 256, 0, st>>>(X, T, in(p + 14), in(p + 15), X);
        if (li < 2) {
            const int cp = 50 + 4 * li;
            int Lc = L + 2;
            k_im2col<<<gblk((long)Bn * Lc * 1536), 256, 0, st>>>(X, BG, Bn, L);
            gemm(st, BG, 0, 0, 1536, 1, in(cp + 0), 0, 0, 1536, 1, in(cp + 1), T, 0, 0, 512,
                 Bn * Lc, 512, 1536, 0, 1, 1);
            k_bn_stats<<<512, 256, 0, st>>>(T, W + WO_MEAN, W + WO_MEAN + 512, Bn * Lc);
            k_bn_elu<<<gblk((long)Bn * Lc * 512), 256, 0, st>>>(T, W + WO_MEAN, W + WO_MEAN + 512,
                                                                in(cp + 2), in(cp + 3), Bn * Lc);
            int Lout = (L + 1) / 2 + 1;
            k_maxpool<<<gblk((long)Bn * Lout * 512), 256, 0, st>>>(T, X, Bn, Lc, Lout);
            L = Lout;
        }
    }
    const int Se = L;               // 1026
    const int ntokE = Bn * Se;
    k_layernorm<<<(unsigned)ntokE, 256, 0, st>>>(X, nullptr, in(110), in(111), ENC);

    // ---------------- decoder ----------------
    const int Ld = 1024;
    const int ntokD = Bn * Ld;
    k_posembed<<<gblk((long)Bn * Ld * 512), 256, 0, st>>>(X, Bn, Ld);
    for (int li = 0; li < 2; ++li) {
        const int p = 58 + 26 * li;
        unsigned kk0, kk1; threefry2x32(0u, 42u, 0u, (unsigned)(100 + li), &kk0, &kk1);
        // masked self-attention (mix=True)
        gemm(st, X, 0, 0, 512, 1, in(p + 0), 0, 0, 512, 1, in(p + 1), Q,  0, 0, 512, ntokD, 512, 512, 0, 1, 1);
        gemm(st, X, 0, 0, 512, 1, in(p + 2), 0, 0, 512, 1, in(p + 3), Kb, 0, 0, 512, ntokD, 512, 512, 0, 1, 1);
        gemm(st, X, 0, 0, 512, 1, in(p + 4), 0, 0, 512, 1, in(p + 5), Vb, 0, 0, 512, ntokD, 512, 512, 0, 1, 1);
        prob_attn_run(st, W, Q, Kb, Vb, CTX, Bn, Ld, 1, kk0, kk1);
        k_mix<<<gblk((long)ntokD * 512), 256, 0, st>>>(CTX, T, Bn, Ld);
        gemm(st, T, 0, 0, 512, 1, in(p + 6), 0, 0, 512, 1, in(p + 7), Q, 0, 0, 512, ntokD, 512, 512, 0, 1, 1);
        k_layernorm<<<(unsigned)ntokD, 256, 0, st>>>(X, Q, in(p + 20), in(p + 21), X);
        // full cross-attention
        gemm(st, X,   0, 0, 512, 1, in(p + 8),  0, 0, 512, 1, in(p + 9),  Q,  0, 0, 512, ntokD, 512, 512, 0, 1, 1);
        gemm(st, ENC, 0, 0, 512, 1, in(p + 10), 0, 0, 512, 1, in(p + 11), Kb, 0, 0, 512, ntokE, 512, 512, 0, 1, 1);
        gemm(st, ENC, 0, 0, 512, 1, in(p + 12), 0, 0, 512, 1, in(p + 13), Vb, 0, 0, 512, ntokE, 512, 512, 0, 1, 1);
        // scores[b,h] = Q[b,:,h,:] x K[b,:,h,:]^T    (K = 64)
        gemm(st, Q,  (long)Ld * 512, 64, 512, 1,
                 Kb, (long)Se * 512, 64, 512, 1, nullptr,
                 BG, (long)8 * Ld * Se, (long)Ld * Se, Se,
                 Ld, Se, 64, 0, Bn, 8);
        k_softmax<<<(unsigned)((long)Bn * 8 * Ld), 256, 0, st>>>(BG, Se, 0.125f);
        // ctx[b,:,h,:] = P x V[b,:,h,:]              (K = Se, handles K % 32 != 0)
        gemm(st, BG, (long)8 * Ld * Se, (long)Ld * Se, Se, 1,
                 Vb, (long)Se * 512, 64, 1, 512, nullptr,
                 CTX, (long)Ld * 512, 64, 512,
                 Ld, 64, Se, 0, Bn, 8);
        gemm(st, CTX, 0, 0, 512, 1, in(p + 14), 0, 0, 512, 1, in(p + 15), T, 0, 0, 512, ntokD, 512, 512, 0, 1, 1);
        k_layernorm<<<(unsigned)ntokD, 256, 0, st>>>(X, T, in(p + 22), in(p + 23), X);
        // FFN
        gemm(st, X,  0, 0, 512,  1, in(p + 16), 0, 0, 512,  1, in(p + 17), BG, 0, 0, 2048, ntokD, 2048, 512, 1, 1, 1);
        gemm(st, BG, 0, 0, 2048, 1, in(p + 18), 0, 0, 2048, 1, in(p + 19), T,  0, 0, 512,  ntokD, 512, 2048, 0, 1, 1);
        k_layernorm<<<(unsigned)ntokD, 256, 0, st>>>(X, T, in(p + 24), in(p + 25), X);
    }
    k_layernorm<<<(unsigned)ntokD, 256, 0, st>>>(X, nullptr, in(112), in(113), T);
    k_proj<<<gblk(Bn * 512 * 7), 256, 0, st>>>(T, in(114), in(115), (float*)d_out, Bn);
}
